// InfoNCELoss_65876208386325
// MI455X (gfx1250) — compile-verified, matrix operand order checked by reading, not run on hardware
//
#include <hip/hip_runtime.h>
#include <hip/hip_bf16.h>

#define NEG_INF (-3.402823466e38f)

typedef __attribute__((ext_vector_type(16))) __bf16 v16bf;
typedef __attribute__((ext_vector_type(8)))  __bf16 v8bf;
typedef __attribute__((ext_vector_type(8)))  float  v8f;

static constexpr int B_SZ  = 4096;
static constexpr int D_SZ  = 256;
static constexpr int TILE  = 128;   // block tile (M and N)
static constexpr int KSLICE = 64;   // K staged per LDS slice

// Wait for all outstanding async (global<->LDS) transfers from this wave.
__device__ __forceinline__ void wait_asynccnt0() {
#if __has_builtin(__builtin_amdgcn_s_wait_asynccnt)
    __builtin_amdgcn_s_wait_asynccnt(0);
#else
    asm volatile("s_wait_asynccnt 0x0" ::: "memory");
#endif
}

// Async copy of 16 bytes global -> LDS (CDNA5 GLOBAL_LOAD_ASYNC_TO_LDS_B128).
// lds_addr: wave-relative LDS byte address (low 32 bits of a __shared__ ptr).
__device__ __forceinline__ void async_copy_b128(unsigned lds_addr,
                                                const void* gaddr) {
    asm volatile("global_load_async_to_lds_b128 %0, %1, off"
                 :: "v"(lds_addr), "v"((unsigned long long)(size_t)gaddr)
                 : "memory");
}

// ---------------------------------------------------------------------------
// 1) Pack fp32 -> bf16 for q and p
// ---------------------------------------------------------------------------
__global__ void pack_bf16_kernel(const float* __restrict__ q,
                                 const float* __restrict__ p,
                                 __bf16* __restrict__ qb,
                                 __bf16* __restrict__ pb) {
    const int N = B_SZ * D_SZ;
    int idx = blockIdx.x * blockDim.x + threadIdx.x;
    if (idx < N) {
        qb[idx] = (__bf16)q[idx];
    } else {
        int k = idx - N;
        pb[k] = (__bf16)p[k];
    }
}

// ---------------------------------------------------------------------------
// 2) Exclusive prefix sum of n_counts (single block, 1024 threads, 4/thread)
// ---------------------------------------------------------------------------
__global__ void scan_kernel(const int* __restrict__ counts,
                            int* __restrict__ offsets) {
    __shared__ int sbuf[1024];
    int tid = threadIdx.x;
    int base = tid * 4;
    int c0 = counts[base + 0];
    int c1 = counts[base + 1];
    int c2 = counts[base + 2];
    int c3 = counts[base + 3];
    int tot = c0 + c1 + c2 + c3;
    sbuf[tid] = tot;
    __syncthreads();
    for (int off = 1; off < 1024; off <<= 1) {
        int v = (tid >= off) ? sbuf[tid - off] : 0;
        __syncthreads();
        sbuf[tid] += v;
        __syncthreads();
    }
    int excl = sbuf[tid] - tot;
    offsets[base + 0] = excl;
    offsets[base + 1] = excl + c0;
    offsets[base + 2] = excl + c0 + c1;
    offsets[base + 3] = excl + c0 + c1 + c2;
}

// ---------------------------------------------------------------------------
// 3) Ragged negatives: one wave per query, <=16 dot products of length 256.
// ---------------------------------------------------------------------------
__global__ void neg_kernel(const float* __restrict__ q,
                           const float* __restrict__ n,
                           const int* __restrict__ offsets,
                           const int* __restrict__ counts,
                           const float* __restrict__ log_tau,
                           float* __restrict__ neg_m,
                           float* __restrict__ neg_s) {
    int tid  = threadIdx.x;
    int lane = tid & 31;
    int w    = tid >> 5;
    int qi   = blockIdx.x * 8 + w;

    float lt = log_tau[0];
    float inv_tau = 1.0f / fminf(fmaxf(__expf(lt), 1e-4f), 1.0f);

    const float4* qr = (const float4*)(q + (size_t)qi * D_SZ);
    float4 a0 = qr[lane * 2 + 0];
    float4 a1 = qr[lane * 2 + 1];

    int cnt = counts[qi];
    int off = offsets[qi];

    float myval = NEG_INF;
    for (int j = 0; j < cnt; ++j) {
        const float4* nr = (const float4*)(n + (size_t)(off + j) * D_SZ);
        float4 b0 = nr[lane * 2 + 0];
        float4 b1 = nr[lane * 2 + 1];
        float d = a0.x * b0.x + a0.y * b0.y + a0.z * b0.z + a0.w * b0.w
                + a1.x * b1.x + a1.y * b1.y + a1.z * b1.z + a1.w * b1.w;
        #pragma unroll
        for (int dd = 1; dd < 32; dd <<= 1) d += __shfl_xor(d, dd, 32);
        d *= inv_tau;
        if (lane == j) myval = d;   // keep logit j in lane j
    }
    float v = (lane < cnt) ? myval : NEG_INF;
    float m = v;
    #pragma unroll
    for (int dd = 1; dd < 32; dd <<= 1) m = fmaxf(m, __shfl_xor(m, dd, 32));
    float e = (lane < cnt) ? __expf(v - m) : 0.0f;
    #pragma unroll
    for (int dd = 1; dd < 32; dd <<= 1) e += __shfl_xor(e, dd, 32);
    if (lane == 0) { neg_m[qi] = m; neg_s[qi] = e; }
}

// ---------------------------------------------------------------------------
// 4) Fused WMMA GEMM: sim tile = (q_tile @ p_tile^T) * inv_tau, folded into
//    per-row and per-column (max, sumexp) partials. Staging uses CDNA5
//    async global->LDS copies (ASYNCcnt), fragments via ds_load_b128,
//    math via v_wmma_f32_16x16x32_bf16.
// ---------------------------------------------------------------------------
__global__ __launch_bounds__(256)
void gemm_lse_kernel(const __bf16* __restrict__ qb,
                     const __bf16* __restrict__ pb,
                     const float* __restrict__ log_tau,
                     float* __restrict__ row_part,   // [32][4096][2]
                     float* __restrict__ col_part,   // [32][4096][2]
                     float* __restrict__ diag) {     // [4096]
    __shared__ __bf16 qs[TILE][KSLICE + 8];
    __shared__ __bf16 ps[TILE][KSLICE + 8];
    __shared__ float rowp_m[2][TILE];
    __shared__ float rowp_s[2][TILE];
    __shared__ float colp_m[4][TILE];
    __shared__ float colp_s[4][TILE];

    const int tid  = threadIdx.x;
    const int lane = tid & 31;
    const int w    = tid >> 5;
    const int wm   = w & 3;    // 0..3 : 32-row band
    const int wn   = w >> 2;   // 0..1 : 64-col band

    const int by = blockIdx.y;          // M tile
    const int bx = blockIdx.x;          // N tile
    const int row0 = by * TILE;
    const int col0 = bx * TILE;

    float lt = log_tau[0];
    float inv_tau = 1.0f / fminf(fmaxf(__expf(lt), 1e-4f), 1.0f);

    v8f acc[2][4] = {};

    const int rsel = lane & 15;
    const int khalf = (lane >> 4) << 3;   // 0 or 8

    for (int ko = 0; ko < D_SZ / KSLICE; ++ko) {
        // ---- async-stage K slice of q and p tiles into LDS ----
        for (int c = tid; c < TILE * (KSLICE / 8); c += 256) {
            int r  = c >> 3;
            int kc = (c & 7) << 3;
            const __bf16* gq = qb + (size_t)(row0 + r) * D_SZ + ko * KSLICE + kc;
            const __bf16* gp = pb + (size_t)(col0 + r) * D_SZ + ko * KSLICE + kc;
            async_copy_b128((unsigned)(size_t)&qs[r][kc], gq);
            async_copy_b128((unsigned)(size_t)&ps[r][kc], gp);
        }
        wait_asynccnt0();
        __syncthreads();

        // ---- 2 WMMA K-steps of 32 per slice ----
        #pragma unroll
        for (int kk = 0; kk < KSLICE; kk += 32) {
            const int kb = kk + khalf;
            v16bf afrag[2], bfrag[4];
            #pragma unroll
            for (int mi = 0; mi < 2; ++mi) {
                int row = wm * 32 + mi * 16 + rsel;
                v8bf lo = *(const v8bf*)&qs[row][kb];
                v8bf hi = *(const v8bf*)&qs[row][kb + 16];
                afrag[mi] = __builtin_shufflevector(lo, hi,
                    0,1,2,3,4,5,6,7,8,9,10,11,12,13,14,15);
            }
            #pragma unroll
            for (int ni = 0; ni < 4; ++ni) {
                int col = wn * 64 + ni * 16 + rsel;
                v8bf lo = *(const v8bf*)&ps[col][kb];
                v8bf hi = *(const v8bf*)&ps[col][kb + 16];
                bfrag[ni] = __builtin_shufflevector(lo, hi,
                    0,1,2,3,4,5,6,7,8,9,10,11,12,13,14,15);
            }
            #pragma unroll
            for (int mi = 0; mi < 2; ++mi)
                #pragma unroll
                for (int ni = 0; ni < 4; ++ni)
                    acc[mi][ni] = __builtin_amdgcn_wmma_f32_16x16x32_bf16(
                        false, afrag[mi], false, bfrag[ni],
                        (short)0, acc[mi][ni], false, false);
        }
        __syncthreads();
    }

    // ---- scale by 1/tau ----
    #pragma unroll
    for (int mi = 0; mi < 2; ++mi)
        #pragma unroll
        for (int ni = 0; ni < 4; ++ni)
            #pragma unroll
            for (int r = 0; r < 8; ++r)
                acc[mi][ni][r] *= inv_tau;

    // ---- diagonal extraction (sim[i,i]) ----
    if (bx == by) {
        #pragma unroll
        for (int mi = 0; mi < 2; ++mi)
            #pragma unroll
            for (int ni = 0; ni < 4; ++ni)
                #pragma unroll
                for (int r = 0; r < 8; ++r) {
                    int lm = wm * 32 + mi * 16 + khalf + r;
                    int ln = wn * 64 + ni * 16 + rsel;
                    if (lm == ln) diag[row0 + lm] = acc[mi][ni][r];
                }
    }

    // ---- per-row (max, sumexp) over this wave's 64 columns ----
    #pragma unroll
    for (int mi = 0; mi < 2; ++mi) {
        #pragma unroll
        for (int r = 0; r < 8; ++r) {
            float m = NEG_INF;
            #pragma unroll
            for (int ni = 0; ni < 4; ++ni) m = fmaxf(m, acc[mi][ni][r]);
            #pragma unroll
            for (int d = 1; d < 16; d <<= 1) m = fmaxf(m, __shfl_xor(m, d, 32));
            float s = 0.0f;
            #pragma unroll
            for (int ni = 0; ni < 4; ++ni) s += __expf(acc[mi][ni][r] - m);
            #pragma unroll
            for (int d = 1; d < 16; d <<= 1) s += __shfl_xor(s, d, 32);
            int lrow = wm * 32 + mi * 16 + khalf + r;
            if ((lane & 15) == 0) { rowp_m[wn][lrow] = m; rowp_s[wn][lrow] = s; }
        }
    }

    // ---- per-column (max, sumexp) over this wave's 32 rows ----
    #pragma unroll
    for (int ni = 0; ni < 4; ++ni) {
        float tm[2], ts[2];
        #pragma unroll
        for (int mi = 0; mi < 2; ++mi) {
            float m = NEG_INF;
            #pragma unroll
            for (int r = 0; r < 8; ++r) m = fmaxf(m, acc[mi][ni][r]);
            m = fmaxf(m, __shfl_xor(m, 16, 32));
            float s = 0.0f;
            #pragma unroll
            for (int r = 0; r < 8; ++r) s += __expf(acc[mi][ni][r] - m);
            s += __shfl_xor(s, 16, 32);
            tm[mi] = m; ts[mi] = s;
        }
        float M = fmaxf(tm[0], tm[1]);
        float S = ts[0] * __expf(tm[0] - M) + ts[1] * __expf(tm[1] - M);
        if (lane < 16) {
            colp_m[wm][wn * 64 + ni * 16 + lane] = M;
            colp_s[wm][wn * 64 + ni * 16 + lane] = S;
        }
    }
    __syncthreads();

    // ---- combine across waves and write global partials ----
    if (tid < TILE) {
        float m0 = rowp_m[0][tid], s0 = rowp_s[0][tid];
        float m1 = rowp_m[1][tid], s1 = rowp_s[1][tid];
        float M = fmaxf(m0, m1);
        float S = s0 * __expf(m0 - M) + s1 * __expf(m1 - M);
        size_t o = ((size_t)bx * B_SZ + row0 + tid) * 2;
        row_part[o] = M; row_part[o + 1] = S;

        float CM = colp_m[0][tid], CS = colp_s[0][tid];
        #pragma unroll
        for (int t = 1; t < 4; ++t) {
            float m = colp_m[t][tid], s = colp_s[t][tid];
            if (m > CM) { CS = CS * __expf(CM - m) + s; CM = m; }
            else        { CS = CS + s * __expf(m - CM); }
        }
        o = ((size_t)by * B_SZ + col0 + tid) * 2;
        col_part[o] = CM; col_part[o + 1] = CS;
    }
}

// ---------------------------------------------------------------------------
// 5) Per-row / per-column finalize
// ---------------------------------------------------------------------------
__global__ void finalize_kernel(const float* __restrict__ row_part,
                                const float* __restrict__ col_part,
                                const float* __restrict__ neg_m,
                                const float* __restrict__ neg_s,
                                const float* __restrict__ diag,
                                float* __restrict__ losses) {  // [8192]
    int idx = blockIdx.x * blockDim.x + threadIdx.x;
    if (idx < B_SZ) {
        int i = idx;
        float M = neg_m[i], S = neg_s[i];
        for (int t = 0; t < 32; ++t) {
            size_t o = ((size_t)t * B_SZ + i) * 2;
            float m = row_part[o], s = row_part[o + 1];
            if (m > M) { S = S * __expf(M - m) + s; M = m; }
            else       { S = S + s * __expf(m - M); }
        }
        losses[i] = (M + logf(S)) - diag[i];
    } else if (idx < 2 * B_SZ) {
        int j = idx - B_SZ;
        float M = NEG_INF, S = 0.0f;
        for (int t = 0; t < 32; ++t) {
            size_t o = ((size_t)t * B_SZ + j) * 2;
            float m = col_part[o], s = col_part[o + 1];
            if (m > M) { S = S * __expf(M - m) + s; M = m; }
            else       { S = S + s * __expf(m - M); }
        }
        losses[idx] = (M + logf(S)) - diag[j];
    }
}

// ---------------------------------------------------------------------------
// 6) Final scalar reduction: (mean(row_loss) + mean(col_loss)) / 2
// ---------------------------------------------------------------------------
__global__ void reduce_kernel(const float* __restrict__ losses,
                              float* __restrict__ out) {
    __shared__ float sb[1024];
    int tid = threadIdx.x;
    float s = 0.0f;
    for (int i = tid; i < 2 * B_SZ; i += 1024) s += losses[i];
    sb[tid] = s;
    __syncthreads();
    for (int off = 512; off > 0; off >>= 1) {
        if (tid < off) sb[tid] += sb[tid + off];
        __syncthreads();
    }
    if (tid == 0) out[0] = sb[0] / (2.0f * (float)B_SZ);
}

// ---------------------------------------------------------------------------
extern "C" void kernel_launch(void* const* d_in, const int* in_sizes, int n_in,
                              void* d_out, int out_size, void* d_ws, size_t ws_size,
                              hipStream_t stream) {
    const float* q_emb   = (const float*)d_in[0];
    const float* p_emb   = (const float*)d_in[1];
    const float* n_emb   = (const float*)d_in[2];
    const int*   n_cnts  = (const int*)d_in[3];
    const float* log_tau = (const float*)d_in[4];
    float* out = (float*)d_out;

    char* w = (char*)d_ws;
    int*   offsets  = (int*)w;                 w += (size_t)B_SZ * sizeof(int);
    float* neg_m    = (float*)w;               w += (size_t)B_SZ * sizeof(float);
    float* neg_s    = (float*)w;               w += (size_t)B_SZ * sizeof(float);
    float* diag     = (float*)w;               w += (size_t)B_SZ * sizeof(float);
    float* losses   = (float*)w;               w += (size_t)2 * B_SZ * sizeof(float);
    float* row_part = (float*)w;               w += (size_t)32 * B_SZ * 2 * sizeof(float);
    float* col_part = (float*)w;               w += (size_t)32 * B_SZ * 2 * sizeof(float);
    __bf16* qb      = (__bf16*)w;              w += (size_t)B_SZ * D_SZ * sizeof(__bf16);
    __bf16* pb      = (__bf16*)w;              w += (size_t)B_SZ * D_SZ * sizeof(__bf16);

    pack_bf16_kernel<<<(2 * B_SZ * D_SZ) / 256, 256, 0, stream>>>(q_emb, p_emb, qb, pb);
    scan_kernel<<<1, 1024, 0, stream>>>(n_cnts, offsets);
    neg_kernel<<<B_SZ / 8, 256, 0, stream>>>(q_emb, n_emb, offsets, n_cnts,
                                             log_tau, neg_m, neg_s);
    dim3 grid(B_SZ / TILE, B_SZ / TILE);
    gemm_lse_kernel<<<grid, 256, 0, stream>>>(qb, pb, log_tau,
                                              row_part, col_part, diag);
    finalize_kernel<<<(2 * B_SZ) / 256, 256, 0, stream>>>(row_part, col_part,
                                                          neg_m, neg_s, diag, losses);
    reduce_kernel<<<1, 1024, 0, stream>>>(losses, out);
}
